// Encoder_41721312313996
// MI455X (gfx1250) — compile-verified, matrix operand order checked by reading, not run on hardware
//
#include <hip/hip_runtime.h>

// ---------------------------------------------------------------------------
// MI455X (gfx1250) fused encoder.
// Bandwidth-bound (~86 GFLOP vs ~30MB resident activations, 23.3 TB/s HBM):
//  - bf16 WMMA (v_wmma_f32_16x16x32_bf16) with f32 accumulation
//  - LN fused into bf16 down-convert
//  - QKV GEMM: one workgroup covers all 768 output cols per 16-row tile;
//    A-fragments register-resident (loaded once from LDS) -> 6x less
//    activation traffic, 48 WMMAs/wave
//  - MLP: both GEMMs fused in one kernel via LDS bf16 intermediate
//  - A-tiles staged with GLOBAL_LOAD_ASYNC_TO_LDS_B128 (ASYNCcnt path);
//    builtin confirmed present (param: AS1 pointer to int vector_size(16))
// ---------------------------------------------------------------------------

#define DEV static __device__ __forceinline__

typedef __attribute__((ext_vector_type(16))) __bf16 v16bf;
typedef __attribute__((ext_vector_type(8)))  __bf16 v8bf;
typedef __attribute__((ext_vector_type(8)))  float  v8f;
typedef int v4i_g __attribute__((vector_size(16)));   // matches builtin param type

constexpr int B_  = 8;
constexpr int S_  = 4096;
constexpr int D_  = 256;
constexpr int L_  = 4;
constexpr int BS_ = B_ * S_;            // 32768 token rows
constexpr float EPS_   = 1e-5f;
constexpr float SCALE_ = 0.015625f;     // 1/sqrt(4096)

DEV unsigned short f2bf(float f) {
  unsigned u = __builtin_bit_cast(unsigned, f);
  u += 0x7FFFu + ((u >> 16) & 1u);      // round-to-nearest-even
  return (unsigned short)(u >> 16);
}

DEV float wave_sum(float v) {
#pragma unroll
  for (int off = 16; off > 0; off >>= 1) v += __shfl_xor(v, off, 32);
  return v;
}

DEV v16bf pack16(v8bf lo, v8bf hi) {
  v16bf r;
#pragma unroll
  for (int i = 0; i < 8; ++i) { r[i] = lo[i]; r[8 + i] = hi[i]; }
  return r;
}

DEV float gelu_exact(float x) {
  return 0.5f * x * (1.0f + erff(x * 0.70710678118654752f));
}

// A-fragment (16x32 bf16) per ISA 7.12.2: lane l holds row m=l&15,
// K halves {h..h+7} and {16+h..16+h+7} with h = (l>>4)*8.
// B-fragment (32x16 bf16): lane l holds col n=l&15, same K striping.
DEV v16bf load_frag(const unsigned short* rowbase, int k0, int khalf) {
  v8bf lo = *reinterpret_cast<const v8bf*>(rowbase + k0 + khalf);
  v8bf hi = *reinterpret_cast<const v8bf*>(rowbase + k0 + 16 + khalf);
  return pack16(lo, hi);
}

// --- CDNA5 async global->LDS staging (ASYNCcnt path), guarded -------------
DEV void stage_b128(const unsigned short* gsrc, unsigned short* lds_dst) {
#if __has_builtin(__builtin_amdgcn_global_load_async_to_lds_b128)
  __builtin_amdgcn_global_load_async_to_lds_b128(
      (__attribute__((address_space(1))) v4i_g*)gsrc,
      (__attribute__((address_space(3))) v4i_g*)lds_dst, 0, 0);
#else
  *reinterpret_cast<uint4*>(lds_dst) = *reinterpret_cast<const uint4*>(gsrc);
#endif
}

DEV void stage_wait() {
#if __has_builtin(__builtin_amdgcn_global_load_async_to_lds_b128)
#if __has_builtin(__builtin_amdgcn_s_wait_asynccnt)
  __builtin_amdgcn_s_wait_asynccnt(0);
#else
  asm volatile("s_wait_asynccnt 0x0" ::: "memory");
#endif
#endif
}

// --------------------------------------------------------------------------
// Weight conversion: [Wq|Wk|Wv|W1|W2] per layer -> bf16, Linear (out,in) rows.
// --------------------------------------------------------------------------
__global__ void wcvt_kernel(const float* __restrict__ Wq, const float* __restrict__ Wk,
                            const float* __restrict__ Wv, const float* __restrict__ W1,
                            const float* __restrict__ W2, unsigned short* __restrict__ Wbf) {
  const int idx = blockIdx.x * 256 + threadIdx.x;
  const int l = idx / (5 * 65536);
  const int r = idx % (5 * 65536);
  const int p = r / 65536;
  const int e = r % 65536;
  const float* src = (p == 0) ? Wq : (p == 1) ? Wk : (p == 2) ? Wv : (p == 3) ? W1 : W2;
  Wbf[idx] = f2bf(src[(size_t)l * 65536 + e]);
}

// --------------------------------------------------------------------------
// LayerNorm over D=256, one wave (32 lanes x 8 cols) per row, bf16 output.
// --------------------------------------------------------------------------
__global__ void ln_bf16_kernel(const float* __restrict__ x,
                               const float* __restrict__ gamma,
                               const float* __restrict__ beta,
                               unsigned short* __restrict__ y) {
  const int wave = threadIdx.x >> 5, lane = threadIdx.x & 31;
  const size_t row = (size_t)blockIdx.x * 8 + wave;
  const float* xr = x + row * D_;
  const int c0 = lane * 8;
  float v[8];
#pragma unroll
  for (int j = 0; j < 8; ++j) v[j] = xr[c0 + j];
  float s = 0.f, s2 = 0.f;
#pragma unroll
  for (int j = 0; j < 8; ++j) { s += v[j]; s2 += v[j] * v[j]; }
  s = wave_sum(s); s2 = wave_sum(s2);
  const float mu   = s  * (1.0f / D_);
  const float var  = s2 * (1.0f / D_) - mu * mu;
  const float rstd = rsqrtf(var + EPS_);
  union { unsigned short us[8]; uint4 v4; } o;
#pragma unroll
  for (int j = 0; j < 8; ++j) {
    const int c = c0 + j;
    o.us[j] = f2bf((v[j] - mu) * rstd * gamma[c] + beta[c]);
  }
  *reinterpret_cast<uint4*>(y + row * D_ + c0) = o.v4;
}

// --------------------------------------------------------------------------
// QKV projection GEMM: [BS x 256] x [256 x 768]^T  (W rows are Linear (out,in)).
// Block = 256 threads = 8 waves, M-tile = 16. A-tile staged once (async->LDS),
// A-fragments then held in VGPRs; each wave sweeps 6 N-tiles (48 WMMAs).
// --------------------------------------------------------------------------
__global__ void qkv_kernel(const unsigned short* __restrict__ hbf,   // [BS][256]
                           const unsigned short* __restrict__ W,     // [768][256]
                           float* __restrict__ qkv) {                // [BS][768]
  __shared__ __align__(16) unsigned short Atile[16][256];
  const int m0 = blockIdx.x * 16;
  const int t  = threadIdx.x;
  {
    const int row = t >> 4, col = (t & 15) * 16;
    const unsigned short* src = hbf + (size_t)(m0 + row) * D_ + col;
    stage_b128(src,     &Atile[row][col]);
    stage_b128(src + 8, &Atile[row][col + 8]);
  }
  stage_wait();
  __syncthreads();

  const int wave = t >> 5, lane = t & 31;
  const int mrow = lane & 15, khalf = (lane >> 4) * 8;

  v16bf afrag[8];
#pragma unroll
  for (int ks = 0; ks < 8; ++ks)
    afrag[ks] = load_frag(&Atile[mrow][0], ks * 32, khalf);

  const int crow0 = m0 + ((lane >> 4) * 8);
#pragma unroll
  for (int nt = 0; nt < 6; ++nt) {
    const int n = nt * 128 + wave * 16 + (lane & 15);          // 0..767
    const unsigned short* wrow = W + (size_t)n * D_;
    v8f acc = {0.f, 0.f, 0.f, 0.f, 0.f, 0.f, 0.f, 0.f};
#pragma unroll
    for (int ks = 0; ks < 8; ++ks) {
      v16bf b = load_frag(wrow, ks * 32, khalf);
      acc = __builtin_amdgcn_wmma_f32_16x16x32_bf16(false, afrag[ks], false, b,
                                                    (short)0, acc, false, false);
    }
#pragma unroll
    for (int i = 0; i < 8; ++i) qkv[(size_t)(crow0 + i) * 768 + n] = acc[i];
  }
}

// --------------------------------------------------------------------------
// Diagonal attention + residual. One wave per token.
// diag[b,h,q] = sum_e Q[t, h*32+e] * K[(b*S + e*128 + q/32), h*32 + q%32]
// att[t, h*32+c] = diag*scale*V[t, h*32+c] + cur[t, h*32+c]
// --------------------------------------------------------------------------
__global__ void attn_kernel(const float* __restrict__ qkv,   // [BS][768] = Q|K|V
                            const float* __restrict__ cur,   // [BS][256]
                            float* __restrict__ att) {       // [BS][256]
  const int wave = threadIdx.x >> 5, lane = threadIdx.x & 31;
  const int tkn = blockIdx.x * 8 + wave;
  const int b = tkn >> 12, q = tkn & (S_ - 1);
  const float* Qrow = qkv + (size_t)tkn * 768;
  const size_t krow = (size_t)b * S_ + (size_t)lane * 128 + (q >> 5);
  const float* Kel  = qkv + krow * 768 + 256 + (q & 31);
  float diag[8];
#pragma unroll
  for (int h = 0; h < 8; ++h) {
    float p = Qrow[h * 32 + lane] * Kel[h * 32];
    diag[h] = wave_sum(p) * SCALE_;
  }
  const float* Vrow = Qrow + 512;
  const float* Crow = cur + (size_t)tkn * D_;
  float*       Arow = att + (size_t)tkn * D_;
#pragma unroll
  for (int h = 0; h < 8; ++h) {
    const int c = h * 32 + lane;
    Arow[c] = diag[h] * Vrow[c] + Crow[c];
  }
}

// --------------------------------------------------------------------------
// Fused MLP: out = gelu(gelu(h @ W1^T + b1) @ W2^T + b2) + att
// Block = 256 threads / 8 waves, M-tile 16. Intermediate kept in LDS as bf16.
// A-fragments register-resident for each GEMM. 32 WMMAs per wave.
// --------------------------------------------------------------------------
__global__ void mlp_kernel(const unsigned short* __restrict__ hbf,  // [BS][256] LN'd
                           const unsigned short* __restrict__ W1,   // [256][256]
                           const float* __restrict__ b1,
                           const unsigned short* __restrict__ W2,   // [256][256]
                           const float* __restrict__ b2,
                           const float* __restrict__ att,           // residual
                           float* __restrict__ outp) {              // [BS][256]
  __shared__ __align__(16) unsigned short Atile[16][256];
  __shared__ __align__(16) unsigned short M1[16][256];
  const int m0 = blockIdx.x * 16;
  const int t  = threadIdx.x;
  {
    const int row = t >> 4, col = (t & 15) * 16;
    const unsigned short* src = hbf + (size_t)(m0 + row) * D_ + col;
    stage_b128(src,     &Atile[row][col]);
    stage_b128(src + 8, &Atile[row][col + 8]);
  }
  stage_wait();
  __syncthreads();

  const int wave = t >> 5, lane = t & 31;
  const int mrow = lane & 15, khalf = (lane >> 4) * 8;
  const int crow0 = 8 * (lane >> 4);

  v16bf afrag[8];
#pragma unroll
  for (int ks = 0; ks < 8; ++ks)
    afrag[ks] = load_frag(&Atile[mrow][0], ks * 32, khalf);

  // ---- GEMM1 + bias + gelu -> LDS bf16 -------------------------------------
#pragma unroll
  for (int tt = 0; tt < 2; ++tt) {
    const int n = tt * 128 + wave * 16 + (lane & 15);
    const unsigned short* wrow = W1 + (size_t)n * D_;
    v8f acc = {0.f, 0.f, 0.f, 0.f, 0.f, 0.f, 0.f, 0.f};
#pragma unroll
    for (int ks = 0; ks < 8; ++ks) {
      v16bf b = load_frag(wrow, ks * 32, khalf);
      acc = __builtin_amdgcn_wmma_f32_16x16x32_bf16(false, afrag[ks], false, b,
                                                    (short)0, acc, false, false);
    }
    const float bias = b1[n];
#pragma unroll
    for (int i = 0; i < 8; ++i)
      M1[crow0 + i][n] = f2bf(gelu_exact(acc[i] + bias));
  }
  __syncthreads();

#pragma unroll
  for (int ks = 0; ks < 8; ++ks)
    afrag[ks] = load_frag(&M1[mrow][0], ks * 32, khalf);

  // ---- GEMM2 + bias + gelu + residual -> global f32 ------------------------
#pragma unroll
  for (int tt = 0; tt < 2; ++tt) {
    const int n = tt * 128 + wave * 16 + (lane & 15);
    const unsigned short* wrow = W2 + (size_t)n * D_;
    v8f acc = {0.f, 0.f, 0.f, 0.f, 0.f, 0.f, 0.f, 0.f};
#pragma unroll
    for (int ks = 0; ks < 8; ++ks) {
      v16bf b = load_frag(wrow, ks * 32, khalf);
      acc = __builtin_amdgcn_wmma_f32_16x16x32_bf16(false, afrag[ks], false, b,
                                                    (short)0, acc, false, false);
    }
    const float bias = b2[n];
#pragma unroll
    for (int i = 0; i < 8; ++i) {
      const size_t r = (size_t)(m0 + crow0 + i);
      outp[r * D_ + n] = gelu_exact(acc[i] + bias) + att[r * D_ + n];
    }
  }
}

// --------------------------------------------------------------------------
// Final: LN of token 0 of each batch. One wave per batch row, f32 output.
// --------------------------------------------------------------------------
__global__ void final_ln_kernel(const float* __restrict__ cur,
                                const float* __restrict__ gamma,
                                const float* __restrict__ beta,
                                float* __restrict__ out) {
  const int wave = threadIdx.x >> 5, lane = threadIdx.x & 31;
  const float* xr = cur + (size_t)wave * S_ * D_;   // token 0 of batch `wave`
  const int c0 = lane * 8;
  float v[8];
#pragma unroll
  for (int j = 0; j < 8; ++j) v[j] = xr[c0 + j];
  float s = 0.f, s2 = 0.f;
#pragma unroll
  for (int j = 0; j < 8; ++j) { s += v[j]; s2 += v[j] * v[j]; }
  s = wave_sum(s); s2 = wave_sum(s2);
  const float mu   = s  * (1.0f / D_);
  const float var  = s2 * (1.0f / D_) - mu * mu;
  const float rstd = rsqrtf(var + EPS_);
#pragma unroll
  for (int j = 0; j < 8; ++j) {
    const int c = c0 + j;
    out[wave * D_ + c] = (v[j] - mu) * rstd * gamma[c] + beta[c];
  }
}

// --------------------------------------------------------------------------
extern "C" void kernel_launch(void* const* d_in, const int* in_sizes, int n_in,
                              void* d_out, int out_size, void* d_ws, size_t ws_size,
                              hipStream_t stream) {
  const float* x     = (const float*)d_in[0];
  const float* Wq    = (const float*)d_in[1];
  const float* Wk    = (const float*)d_in[2];
  const float* Wv    = (const float*)d_in[3];
  const float* W1    = (const float*)d_in[4];
  const float* b1    = (const float*)d_in[5];
  const float* W2    = (const float*)d_in[6];
  const float* b2    = (const float*)d_in[7];
  const float* gamma = (const float*)d_in[8];
  const float* beta  = (const float*)d_in[9];
  float* out = (float*)d_out;
  (void)in_sizes; (void)n_in; (void)out_size; (void)ws_size;

  char* ws = (char*)d_ws;
  size_t off = 0;
  auto wsalloc = [&](size_t bytes) -> void* {
    void* p = ws + off;
    off = (off + bytes + 255) & ~(size_t)255;
    return p;
  };
  float* cur = (float*)wsalloc((size_t)BS_ * D_ * 4);          // activations
  float* att = (float*)wsalloc((size_t)BS_ * D_ * 4);          // attended residual
  float* qkv = (float*)wsalloc((size_t)BS_ * 768 * 4);         // Q|K|V
  unsigned short* hbf = (unsigned short*)wsalloc((size_t)BS_ * D_ * 2);
  unsigned short* Wbf = (unsigned short*)wsalloc((size_t)L_ * 5 * 65536 * 2);

  const int wtot = L_ * 5 * 65536;
  wcvt_kernel<<<wtot / 256, 256, 0, stream>>>(Wq, Wk, Wv, W1, W2, Wbf);
  (void)hipMemcpyAsync(cur, x, (size_t)BS_ * D_ * 4, hipMemcpyDeviceToDevice, stream);

  for (int l = 0; l < L_; ++l) {
    const unsigned short* Wl = Wbf + (size_t)l * 5 * 65536;
    ln_bf16_kernel<<<BS_ / 8, 256, 0, stream>>>(cur, gamma, beta, hbf);
    qkv_kernel<<<BS_ / 16, 256, 0, stream>>>(hbf, Wl, qkv);
    attn_kernel<<<BS_ / 8, 256, 0, stream>>>(qkv, cur, att);
    ln_bf16_kernel<<<BS_ / 8, 256, 0, stream>>>(att, gamma, beta, hbf);
    mlp_kernel<<<BS_ / 16, 256, 0, stream>>>(hbf, Wl + 3 * 65536, b1 + l * D_,
                                             Wl + 4 * 65536, b2 + l * D_, att, cur);
  }
  final_ln_kernel<<<1, 256, 0, stream>>>(cur, gamma, beta, out);
}
